// Mamba2Layer_44830868636060
// MI455X (gfx1250) — compile-verified, matrix operand order checked by reading, not run on hardware
//
#include <hip/hip_runtime.h>
#include <stdint.h>
#include <math.h>

// ---------------------------------------------------------------------------
// Mamba2 layer for MI455X (gfx1250): bf16 WMMA GEMMs, async global->LDS
// double-buffered staging, fused elementwise ops.
// ---------------------------------------------------------------------------

typedef __attribute__((ext_vector_type(16))) __bf16 v16bf;
typedef __attribute__((ext_vector_type(8)))  float  v8f;

static constexpr int  BSZ    = 2;
static constexpr int  TLEN   = 2048;
static constexpr int  DMODEL = 1024;
static constexpr int  DINNER = 2048;
static constexpr int  NH     = 32;
static constexpr int  HD     = 64;
static constexpr int  NST    = 64;
static constexpr int  LCH    = 256;
static constexpr int  NC     = 8;
static constexpr int  DPROJ  = 4256;   // 2*2048 + 2*64 + 32
static constexpr float SSD_SCALE = 0.125f; // 64^-0.5

__device__ __forceinline__ unsigned short f32_to_bf16(float f) {
  union { float f; unsigned int u; } v; v.f = f;
  unsigned int u = v.u;
  unsigned int r = u + 0x7FFFu + ((u >> 16) & 1u);
  if ((u & 0x7F800000u) == 0x7F800000u) r = u;   // keep inf/nan
  return (unsigned short)(r >> 16);
}
__device__ __forceinline__ float bf16_to_f32(unsigned short h) {
  union { unsigned int u; float f; } v; v.u = ((unsigned int)h) << 16;
  return v.f;
}
__device__ __forceinline__ float siluf(float x) { return x / (1.f + expf(-x)); }

// ---------------------------------------------------------------------------
// Generic batched NT GEMM:  C[m,n] = acc += A[m,k] * B[n,k]   (bf16 x bf16 -> f32)
// Block tile 128x64, 4 waves, each wave 32x64 (2 A-frags x 4 B-frags, 8 WMMA).
// Tiles staged with GLOBAL_LOAD_ASYNC_TO_LDS_B128 (ASYNCcnt), double-buffered
// so tile k+1 transfers overlap the 8 WMMAs of tile k. OOB tile rows are
// CLAMPED (not predicated): garbage only feeds store-guarded output rows/cols.
// ---------------------------------------------------------------------------
struct GemmArgs {
  const unsigned short* A;
  const unsigned short* B;
  void*       C;            // float* or ushort* (outBf16)
  const float* rowscale;    // nullable
  int M, N, K;
  int lda, ldb, ldc;
  long long sA0, sA1, sA2;
  long long sB0, sB1, sB2;
  long long sC0, sC1, sC2;
  long long sR0, sR1, sR2;
  int D1, D2;
  int rsRowStride;
  float alpha;
  int accumulate;
  int outBf16;
};

// LDS geometry: row pitch 20 dwords (80B) -> conflict-free b128 fragment reads,
// A tile 128 rows + B tile 64 rows per buffer, two buffers.
static constexpr int LDS_PITCH   = 20;                     // dwords per row
static constexpr int LDS_BUF_DW  = (128 + 64) * LDS_PITCH; // dwords per buffer

// Issue one 64xK32 (B) / 128xK32 (A) tile transfer via async-to-LDS.
__device__ __forceinline__ void stage_async_tile(
    const unsigned short* __restrict__ A, const unsigned short* __restrict__ B,
    int lda, int ldb, int M, int N, int tileM, int tileN, int k0,
    unsigned ldsA_byte, unsigned ldsB_byte, int tid) {
#pragma unroll
  for (int rep = 0; rep < 4; ++rep) {
    const int chunk = tid + rep * 128;           // 0..511
    const int row   = chunk >> 2;
    const int cq    = chunk & 3;
    const int ar    = min(tileM + row, M - 1);
    const unsigned short* gp = A + (long long)ar * lda + k0 + cq * 8;
    const unsigned la = ldsA_byte + (unsigned)((row * LDS_PITCH + cq * 4) * 4);
    asm volatile("global_load_async_to_lds_b128 %0, %1, off"
                 :: "v"(la), "v"(gp) : "memory");
  }
#pragma unroll
  for (int rep = 0; rep < 2; ++rep) {
    const int chunk = tid + rep * 128;           // 0..255
    const int row   = chunk >> 2;
    const int cq    = chunk & 3;
    const int br    = min(tileN + row, N - 1);
    const unsigned short* gp = B + (long long)br * ldb + k0 + cq * 8;
    const unsigned lb = ldsB_byte + (unsigned)((row * LDS_PITCH + cq * 4) * 4);
    asm volatile("global_load_async_to_lds_b128 %0, %1, off"
                 :: "v"(lb), "v"(gp) : "memory");
  }
}

__global__ __launch_bounds__(128) void gemm_nt_bf16(GemmArgs g) {
  __shared__ alignas(16) unsigned int lds[2 * LDS_BUF_DW];

  const int z  = blockIdx.z;
  const int d2 = z % g.D2;
  const int zt = z / g.D2;
  const int d1 = zt % g.D1;
  const int d0 = zt / g.D1;

  const unsigned short* A = g.A + (long long)d0 * g.sA0 + (long long)d1 * g.sA1 + (long long)d2 * g.sA2;
  const unsigned short* B = g.B + (long long)d0 * g.sB0 + (long long)d1 * g.sB1 + (long long)d2 * g.sB2;
  const long long cOff = (long long)d0 * g.sC0 + (long long)d1 * g.sC1 + (long long)d2 * g.sC2;
  const float* rs = g.rowscale
      ? (g.rowscale + (long long)d0 * g.sR0 + (long long)d1 * g.sR1 + (long long)d2 * g.sR2)
      : nullptr;

  const int tileN = blockIdx.x * 64;
  const int tileM = blockIdx.y * 128;
  const int tid   = threadIdx.x;
  const int lane  = tid & 31;
  const int wave  = tid >> 5;
  const int halfl = lane >> 4;     // 0: lanes 0-15, 1: lanes 16-31
  const int lrow  = lane & 15;

  // LDS byte addresses for the async DMA (low 32 bits of generic LDS pointer
  // == LDS byte offset per the flat-address aperture mapping).
  const unsigned ldsBase = (unsigned)(size_t)(&lds[0]);
  const unsigned bufBytes = (unsigned)(LDS_BUF_DW * 4);
  const unsigned ldsAB[2] = { ldsBase, ldsBase + bufBytes };
  const unsigned ldsBB[2] = { ldsBase + (unsigned)(128 * LDS_PITCH * 4),
                              ldsBase + bufBytes + (unsigned)(128 * LDS_PITCH * 4) };

  v8f acc[2][4] = {};

  // prologue: stage tile 0 into buffer 0
  stage_async_tile(A, B, g.lda, g.ldb, g.M, g.N, tileM, tileN, 0,
                   ldsAB[0], ldsBB[0], tid);
  asm volatile("s_wait_asynccnt 0x0" ::: "memory");
  __syncthreads();

  int buf = 0;
  for (int k0 = 0; k0 < g.K; k0 += 32, buf ^= 1) {
    // kick off next tile into the other buffer (overlaps with WMMAs below)
    if (k0 + 32 < g.K) {
      stage_async_tile(A, B, g.lda, g.ldb, g.M, g.N, tileM, tileN, k0 + 32,
                       ldsAB[buf ^ 1], ldsBB[buf ^ 1], tid);
      // L2 prefetch two tiles ahead
      if (k0 + 64 < g.K) {
        const int ar = min(tileM + tid, g.M - 1);
        __builtin_prefetch(A + (long long)ar * g.lda + k0 + 64, 0, 0);
        if (tid < 64) {
          const int br = min(tileN + tid, g.N - 1);
          __builtin_prefetch(B + (long long)br * g.ldb + k0 + 64, 0, 0);
        }
      }
    }

    const unsigned int* ldsA = lds + buf * LDS_BUF_DW;
    const unsigned int* ldsB = ldsA + 128 * LDS_PITCH;

    // --- build fragments per documented CDNA5 16-bit A/B layout:
    // lane: row = lane%16, K-halves split by lane/16; VGPR0-3 = K[half*8..+8),
    // VGPR4-7 = K[16+half*8..+8).  Two b128 LDS reads per fragment.
    union FragU { v16bf v; uint4 q[2]; };
    FragU afr[2];
#pragma unroll
    for (int s = 0; s < 2; ++s) {
      const unsigned int* ab = ldsA + (wave * 32 + s * 16 + lrow) * LDS_PITCH + halfl * 4;
      afr[s].q[0] = *(const uint4*)(ab);
      afr[s].q[1] = *(const uint4*)(ab + 8);
    }
    FragU bfr[4];
#pragma unroll
    for (int j = 0; j < 4; ++j) {
      const unsigned int* bb = ldsB + (j * 16 + lrow) * LDS_PITCH + halfl * 4;
      bfr[j].q[0] = *(const uint4*)(bb);
      bfr[j].q[1] = *(const uint4*)(bb + 8);
    }
#pragma unroll
    for (int s = 0; s < 2; ++s) {
#pragma unroll
      for (int j = 0; j < 4; ++j) {
        acc[s][j] = __builtin_amdgcn_wmma_f32_16x16x32_bf16(
            false, afr[s].v, false, bfr[j].v, (short)0, acc[s][j], false, false);
      }
    }
    // next buffer fully written + everyone done reading this buffer
    asm volatile("s_wait_asynccnt 0x0" ::: "memory");
    __syncthreads();
  }

  // --- epilogue: C/D layout -> lane l: col = l%16; VGPR r: row = r + 8*(l/16)
#pragma unroll
  for (int s = 0; s < 2; ++s) {
    const int rowBase = tileM + wave * 32 + s * 16 + 8 * halfl;
#pragma unroll
    for (int j = 0; j < 4; ++j) {
      const int col = tileN + j * 16 + lrow;
      if (col >= g.N) continue;
#pragma unroll
      for (int r = 0; r < 8; ++r) {
        const int row = rowBase + r;
        if (row >= g.M) continue;
        float v = acc[s][j][r] * g.alpha;
        if (rs) v *= rs[(long long)row * g.rsRowStride];
        const long long cidx = cOff + (long long)row * g.ldc + col;
        if (g.outBf16) {
          ((unsigned short*)g.C)[cidx] = f32_to_bf16(v);
        } else {
          float* cp = (float*)g.C;
          if (g.accumulate) v += cp[cidx];
          cp[cidx] = v;
        }
      }
    }
  }
}

// ---------------------------------------------------------------------------
// Elementwise / small kernels
// ---------------------------------------------------------------------------
__global__ void cast_f32_bf16_kernel(const float* __restrict__ src,
                                     unsigned short* __restrict__ dst, long long n) {
  long long i = (long long)blockIdx.x * blockDim.x + threadIdx.x;
  if (i < n) dst[i] = f32_to_bf16(src[i]);
}

// per (b,c,h): dt = clip(softplus(dt_raw+bias)), cum = cumsum(A*dt),
// chunk_decay, decay_to_end, exp(cum)*scale
__global__ void dtcum_kernel(const unsigned short* __restrict__ projbf,
                             const float* __restrict__ A_log,
                             const float* __restrict__ dt_bias,
                             float* __restrict__ dt, float* __restrict__ cum,
                             float* __restrict__ d2e, float* __restrict__ dfsS,
                             float* __restrict__ cdec) {
  int idx = blockIdx.x * blockDim.x + threadIdx.x;
  if (idx >= BSZ * NC * NH) return;
  const int h = idx & 31;
  const int c = (idx >> 5) & 7;
  const int b = idx >> 8;
  const float Ah = -expf(A_log[h]);
  const float bias = dt_bias[h];
  float run = 0.f;
  for (int l = 0; l < LCH; ++l) {
    const int t = c * LCH + l;
    float raw = bf16_to_f32(projbf[(long long)(b * TLEN + t) * DPROJ + 4224 + h]) + bias;
    float sp = (raw > 20.f) ? raw : log1pf(expf(raw));
    sp = fminf(fmaxf(sp, 1e-4f), 0.5f);
    dt[(long long)(b * TLEN + t) * NH + h] = sp;
    run += Ah * sp;
    cum[((long long)(b * NC + c) * LCH + l) * NH + h] = run;
  }
  cdec[idx] = expf(run);
  const float last = run;
  for (int l = 0; l < LCH; ++l) {
    const long long o = ((long long)(b * NC + c) * LCH + l) * NH + h;
    const float cv = cum[o];
    d2e[o]  = expf(last - cv);
    dfsS[o] = expf(cv) * SSD_SCALE;
  }
}

// cast B/C groups out of proj; also B transposed to (n,l) for delta_h GEMM
__global__ void castbc_kernel(const unsigned short* __restrict__ projbf,
                              unsigned short* __restrict__ Bbf,
                              unsigned short* __restrict__ Cbf,
                              unsigned short* __restrict__ Btr) {
  int idx = blockIdx.x * blockDim.x + threadIdx.x;  // (b,t,n)
  if (idx >= BSZ * TLEN * NST) return;
  const int n = idx & 63;
  const int t = (idx >> 6) & 2047;
  const int b = idx >> 17;
  const long long po = (long long)(b * TLEN + t) * DPROJ;
  const unsigned short bv = projbf[po + 4096 + n];
  const unsigned short cv = projbf[po + 4160 + n];
  Bbf[idx] = bv;
  Cbf[idx] = cv;
  const int c = t >> 8, l = t & 255;
  Btr[(((long long)(b * NC + c) * NST) + n) * LCH + l] = bv;
}

// depthwise causal conv(4) + SiLU, fused *dt and transpose into (b,c,h,p,l)
__global__ void conv_kernel(const unsigned short* __restrict__ projbf,
                            const float* __restrict__ conv_w,
                            const float* __restrict__ dt,
                            const float* __restrict__ d2e,
                            unsigned short* __restrict__ xdt_tr,
                            unsigned short* __restrict__ xdtD_tr) {
  long long idx = (long long)blockIdx.x * blockDim.x + threadIdx.x;  // (b,t,d)
  if (idx >= (long long)BSZ * TLEN * DINNER) return;
  const int d = (int)(idx & 2047);
  const int t = (int)((idx >> 11) & 2047);
  const int b = (int)(idx >> 22);
  float acc = 0.f;
#pragma unroll
  for (int k = 0; k < 4; ++k) {
    const int tk = t - 3 + k;
    if (tk >= 0)
      acc += conv_w[d * 4 + k] *
             bf16_to_f32(projbf[(long long)(b * TLEN + tk) * DPROJ + DINNER + d]);
  }
  const float s = siluf(acc);
  const int h = d >> 6, p = d & 63, c = t >> 8, l = t & 255;
  const float xv = s * dt[(long long)(b * TLEN + t) * NH + h];
  const long long o = ((((long long)(b * NC + c) * NH + h) * HD + p) * LCH + l);
  xdt_tr[o] = f32_to_bf16(xv);
  const float de = d2e[((long long)(b * NC + c) * LCH + l) * NH + h];
  xdtD_tr[o] = f32_to_bf16(xv * de);
}

// masked decay: Sbf[bc,h,i,j] = (i>=j) ? bf16(score*scale*exp(cum_i - cum_j)) : 0
__global__ void sbf_kernel(const float* __restrict__ scores,
                           const float* __restrict__ cum,
                           unsigned short* __restrict__ Sbf) {
  long long idx = (long long)blockIdx.x * blockDim.x + threadIdx.x;
  if (idx >= (long long)BSZ * NC * NH * LCH * LCH) return;
  const int j  = (int)(idx & 255);
  const int i  = (int)((idx >> 8) & 255);
  const int h  = (int)((idx >> 16) & 31);
  const int bc = (int)(idx >> 21);
  unsigned short out = 0;
  if (i >= j) {
    const float sc = scores[((long long)bc * LCH + i) * LCH + j];
    const float ci = cum[((long long)bc * LCH + i) * NH + h];
    const float cj = cum[((long long)bc * LCH + j) * NH + h];
    out = f32_to_bf16(sc * SSD_SCALE * expf(ci - cj));
  }
  Sbf[idx] = out;
}

// sequential scan over 8 chunks; states[c] = state entering chunk c (bf16)
__global__ void scan_kernel(const float* __restrict__ delta,
                            const float* __restrict__ cdec,
                            unsigned short* __restrict__ states) {
  int idx = blockIdx.x * blockDim.x + threadIdx.x;  // (b,h,p,n)
  if (idx >= BSZ * NH * HD * NST) return;
  const int n = idx & 63;
  const int p = (idx >> 6) & 63;
  const int h = (idx >> 12) & 31;
  const int b = idx >> 17;
  float s = 0.f;
#pragma unroll
  for (int c = 0; c < NC; ++c) {
    const long long o = ((((long long)(b * NC + c) * NH + h) * HD + p) * NST + n);
    states[o] = f32_to_bf16(s);
    s = cdec[(b * NC + c) * NH + h] * s + delta[o];
  }
}

// RMS norm over d_inner + silu(z) gate -> bf16 for final GEMM
__global__ __launch_bounds__(256) void rmsgate_kernel(
    const float* __restrict__ y, const unsigned short* __restrict__ projbf,
    unsigned short* __restrict__ ygbf) {
  const int bt = blockIdx.x;                 // 0..4095
  const float* yrow = y + (long long)bt * DINNER;
  float ss = 0.f;
  for (int i = threadIdx.x; i < DINNER; i += 256) {
    const float v = yrow[i];
    ss += v * v;
  }
#pragma unroll
  for (int off = 16; off > 0; off >>= 1) ss += __shfl_xor(ss, off, 32);
  __shared__ float red[8];
  if ((threadIdx.x & 31) == 0) red[threadIdx.x >> 5] = ss;
  __syncthreads();
  float tot = 0.f;
#pragma unroll
  for (int w = 0; w < 8; ++w) tot += red[w];
  const float inv = rsqrtf(tot / (float)DINNER + 1.1920928955078125e-7f);
  for (int i = threadIdx.x; i < DINNER; i += 256) {
    const float zv = bf16_to_f32(projbf[(long long)bt * DPROJ + i]);
    ygbf[(long long)bt * DINNER + i] = f32_to_bf16(yrow[i] * inv * siluf(zv));
  }
}

// ---------------------------------------------------------------------------
// Host orchestration
// ---------------------------------------------------------------------------
static inline dim3 gemm_grid(int M, int N, int batches) {
  return dim3((unsigned)((N + 63) / 64), (unsigned)((M + 127) / 128), (unsigned)batches);
}

extern "C" void kernel_launch(void* const* d_in, const int* in_sizes, int n_in,
                              void* d_out, int out_size, void* d_ws, size_t ws_size,
                              hipStream_t stream) {
  const float* x       = (const float*)d_in[0];
  const float* W_in    = (const float*)d_in[1];
  const float* conv_w  = (const float*)d_in[2];
  const float* A_log   = (const float*)d_in[3];
  const float* dt_bias = (const float*)d_in[4];
  const float* W_out   = (const float*)d_in[5];
  float* out = (float*)d_out;

  const long long BT = (long long)BSZ * TLEN;             // 4096
  char* w = (char*)d_ws;
  auto carve = [&](long long bytes) {
    char* p = w;
    w += (bytes + 255) & ~(long long)255;
    return p;
  };
  unsigned short* projbf  = (unsigned short*)carve(BT * DPROJ * 2);            // 34.9 MB
  unsigned short* xbf     = (unsigned short*)carve(BT * DMODEL * 2);           // 8.4 MB
  unsigned short* wibf    = (unsigned short*)carve((long long)DPROJ * DMODEL * 2);
  unsigned short* wobf    = (unsigned short*)carve((long long)DMODEL * DINNER * 2);
  float*          dt      = (float*)carve(BT * NH * 4);
  float*          cum     = (float*)carve(BT * NH * 4);
  float*          d2e     = (float*)carve(BT * NH * 4);
  float*          dfsS    = (float*)carve(BT * NH * 4);
  float*          cdec    = (float*)carve((long long)BSZ * NC * NH * 4);
  unsigned short* Bbf     = (unsigned short*)carve(BT * NST * 2);
  unsigned short* Cbf     = (unsigned short*)carve(BT * NST * 2);
  unsigned short* Btr     = (unsigned short*)carve(BT * NST * 2);
  unsigned short* xdt_tr  = (unsigned short*)carve(BT * DINNER * 2);           // 16.8 MB
  unsigned short* xdtD_tr = (unsigned short*)carve(BT * DINNER * 2);           // 16.8 MB
  float*          scores  = (float*)carve((long long)BSZ * NC * LCH * LCH * 4);// 4.2 MB
  unsigned short* Sbf     = (unsigned short*)carve((long long)BSZ * NC * NH * LCH * LCH * 2); // 67 MB
  float*          deltah  = (float*)carve((long long)BSZ * NC * NH * HD * NST * 4);           // 8.4 MB
  unsigned short* states  = (unsigned short*)carve((long long)BSZ * NC * NH * HD * NST * 2);  // 4.2 MB
  float*          ybuf    = (float*)carve(BT * DINNER * 4);                    // 33.6 MB
  unsigned short* ygbf    = (unsigned short*)carve(BT * DINNER * 2);           // 16.8 MB

  // --- casts to bf16
  {
    long long n1 = BT * DMODEL, n2 = (long long)DPROJ * DMODEL, n3 = (long long)DMODEL * DINNER;
    cast_f32_bf16_kernel<<<(unsigned)((n1 + 255) / 256), 256, 0, stream>>>(x, xbf, n1);
    cast_f32_bf16_kernel<<<(unsigned)((n2 + 255) / 256), 256, 0, stream>>>(W_in, wibf, n2);
    cast_f32_bf16_kernel<<<(unsigned)((n3 + 255) / 256), 256, 0, stream>>>(W_out, wobf, n3);
  }

  // --- 1) in-projection: proj = x @ W_in^T  (4096 x 4256 x 1024), bf16 out
  {
    GemmArgs g{};
    g.A = xbf; g.B = wibf; g.C = projbf; g.rowscale = nullptr;
    g.M = (int)BT; g.N = DPROJ; g.K = DMODEL;
    g.lda = DMODEL; g.ldb = DMODEL; g.ldc = DPROJ;
    g.D1 = 1; g.D2 = 1; g.alpha = 1.f; g.accumulate = 0; g.outBf16 = 1;
    gemm_nt_bf16<<<gemm_grid(g.M, g.N, 1), 128, 0, stream>>>(g);
  }

  // --- 2) dt / cumsum / decays
  dtcum_kernel<<<2, 256, 0, stream>>>(projbf, A_log, dt_bias, dt, cum, d2e, dfsS, cdec);

  // --- 3) B/C casts + B transpose
  castbc_kernel<<<(unsigned)((BT * NST + 255) / 256), 256, 0, stream>>>(projbf, Bbf, Cbf, Btr);

  // --- 4) conv + silu + *dt, write transposed bf16 operands
  conv_kernel<<<(unsigned)((BT * DINNER + 255) / 256), 256, 0, stream>>>(
      projbf, conv_w, dt, d2e, xdt_tr, xdtD_tr);

  // --- 5) scores = C @ B^T per (b,c):  256 x 256 x 64
  {
    GemmArgs g{};
    g.A = Cbf; g.B = Bbf; g.C = scores; g.rowscale = nullptr;
    g.M = LCH; g.N = LCH; g.K = NST;
    g.lda = NST; g.ldb = NST; g.ldc = LCH;
    g.sA0 = (long long)LCH * NST; g.sB0 = (long long)LCH * NST; g.sC0 = (long long)LCH * LCH;
    g.D1 = 1; g.D2 = 1; g.alpha = 1.f;
    gemm_nt_bf16<<<gemm_grid(LCH, LCH, BSZ * NC), 128, 0, stream>>>(g);
  }

  // --- 6) masked decay scores -> bf16 per head
  sbf_kernel<<<(unsigned)(((long long)BSZ * NC * NH * LCH * LCH + 255) / 256), 256, 0, stream>>>(
      scores, cum, Sbf);

  // --- 7) y_intra = S_h @ xdt_h  per (b,c,h): 256 x 64 x 256
  {
    GemmArgs g{};
    g.A = Sbf; g.B = xdt_tr; g.C = ybuf; g.rowscale = nullptr;
    g.M = LCH; g.N = HD; g.K = LCH;
    g.lda = LCH; g.ldb = LCH; g.ldc = DINNER;
    g.sA0 = (long long)NC * NH * LCH * LCH; g.sA1 = (long long)NH * LCH * LCH; g.sA2 = (long long)LCH * LCH;
    g.sB0 = (long long)NC * NH * HD * LCH;  g.sB1 = (long long)NH * HD * LCH;  g.sB2 = (long long)HD * LCH;
    g.sC0 = (long long)TLEN * DINNER;       g.sC1 = (long long)LCH * DINNER;   g.sC2 = HD;
    g.D1 = NC; g.D2 = NH; g.alpha = 1.f; g.accumulate = 0; g.outBf16 = 0;
    gemm_nt_bf16<<<gemm_grid(LCH, HD, BSZ * NC * NH), 128, 0, stream>>>(g);
  }

  // --- 8) delta_h = (xdt*d2e)^T @ B  per (b,c,h): 64 x 64 x 256
  {
    GemmArgs g{};
    g.A = xdtD_tr; g.B = Btr; g.C = deltah; g.rowscale = nullptr;
    g.M = HD; g.N = NST; g.K = LCH;
    g.lda = LCH; g.ldb = LCH; g.ldc = NST;
    g.sA0 = (long long)NC * NH * HD * LCH; g.sA1 = (long long)NH * HD * LCH; g.sA2 = (long long)HD * LCH;
    g.sB0 = (long long)NC * NST * LCH;     g.sB1 = (long long)NST * LCH;     g.sB2 = 0;
    g.sC0 = (long long)NC * NH * HD * NST; g.sC1 = (long long)NH * HD * NST; g.sC2 = (long long)HD * NST;
    g.D1 = NC; g.D2 = NH; g.alpha = 1.f;
    gemm_nt_bf16<<<gemm_grid(HD, NST, BSZ * NC * NH), 128, 0, stream>>>(g);
  }

  // --- 9) chunk scan -> per-chunk entry states (bf16)
  scan_kernel<<<(unsigned)((BSZ * NH * HD * NST + 255) / 256), 256, 0, stream>>>(
      deltah, cdec, states);

  // --- 10) y_inter = C @ state^T per (b,c,h): 256 x 64 x 64, row-scaled by
  //          exp(cum)*SSD_SCALE, accumulated into ybuf
  {
    GemmArgs g{};
    g.A = Cbf; g.B = states; g.C = ybuf; g.rowscale = dfsS;
    g.M = LCH; g.N = HD; g.K = NST;
    g.lda = NST; g.ldb = NST; g.ldc = DINNER;
    g.sA0 = (long long)NC * LCH * NST;     g.sA1 = (long long)LCH * NST;      g.sA2 = 0;
    g.sB0 = (long long)NC * NH * HD * NST; g.sB1 = (long long)NH * HD * NST;  g.sB2 = (long long)HD * NST;
    g.sC0 = (long long)TLEN * DINNER;      g.sC1 = (long long)LCH * DINNER;   g.sC2 = HD;
    g.sR0 = (long long)NC * LCH * NH;      g.sR1 = (long long)LCH * NH;       g.sR2 = 1;
    g.rsRowStride = NH;
    g.D1 = NC; g.D2 = NH; g.alpha = 1.f; g.accumulate = 1; g.outBf16 = 0;
    gemm_nt_bf16<<<gemm_grid(LCH, HD, BSZ * NC * NH), 128, 0, stream>>>(g);
  }

  // --- 11) RMS norm + silu(z) gate -> bf16
  rmsgate_kernel<<<(unsigned)BT, 256, 0, stream>>>(ybuf, projbf, ygbf);

  // --- 12) out = yg @ W_out^T  (4096 x 1024 x 2048) -> fp32 d_out
  {
    GemmArgs g{};
    g.A = ygbf; g.B = wobf; g.C = out; g.rowscale = nullptr;
    g.M = (int)BT; g.N = DMODEL; g.K = DINNER;
    g.lda = DINNER; g.ldb = DINNER; g.ldc = DMODEL;
    g.D1 = 1; g.D2 = 1; g.alpha = 1.f; g.accumulate = 0; g.outBf16 = 0;
    gemm_nt_bf16<<<gemm_grid(g.M, g.N, 1), 128, 0, stream>>>(g);
  }
  (void)in_sizes; (void)n_in; (void)out_size; (void)ws_size;
}
